// EventValueEmbedding_34102040330711
// MI455X (gfx1250) — matrix operand order repaired and checked by reading, not compiled, not run on hardware
//
#include <hip/hip_runtime.h>

typedef __attribute__((ext_vector_type(16))) _Float16 v16h;
typedef __attribute__((ext_vector_type(8)))  float    v8f;
typedef __attribute__((ext_vector_type(4)))  float    v4f;

#define D_MODEL 128
#define HID 64
#define WAVES_PER_BLOCK 4
#define BLOCK_THREADS 128
#define MTILES 8            // 128 / 16 d_model tiles

// -----------------------------------------------------------------------------
// Transposed orientation:  D[128 x 16tok] = W2^T[128 x 64] (A) x H^T[64 x 16] (B)
//   A (per mtile, per kb): A[m][k] = w2[k][m]  -- loop-invariant, LDS-swizzled once
//   B (per tile):          B[k][n] = relu(v_n * w1[k] + b1[k])  (f16)
//   C/D: lane owns token n = lane&15 and 8 consecutive m values
//        -> branchless select + two global_store_b128 per mtile per lane.
// -----------------------------------------------------------------------------
__global__ __launch_bounds__(BLOCK_THREADS)
void EventValueEmbedding_34102040330711_kernel(
    const int*   __restrict__ variate_ids,   // [B*T]
    const float* __restrict__ value_num,     // [B*T]
    const int*   __restrict__ cat_ids,       // [B*T]
    const int*   __restrict__ variate_type,  // [NUM_VARIATES]
    const float* __restrict__ numeric_means, // [NUM_VARIATES]
    const float* __restrict__ numeric_stds,  // [NUM_VARIATES]
    const float* __restrict__ w1,            // [64]
    const float* __restrict__ b1,            // [64]
    const float* __restrict__ w2,            // [64,128]
    const float* __restrict__ b2,            // [128]
    const float* __restrict__ cat_embed,     // [NUM_CAT,128]
    float*       __restrict__ out,           // [B*T,128]
    int num_tiles, int tiles_per_wave)
{
    // A fragments of w2^T in WMMA A layout: frag = mtile*2 + kb, 32 lanes x 16 halves
    __shared__ __align__(32) _Float16 lds_a[MTILES * 2 * 32 * 16]; // 16 KB
    __shared__ float lds_w1[HID];
    __shared__ float lds_b1[HID];
    __shared__ __align__(16) float lds_b2[D_MODEL];

    const int tid = threadIdx.x;

    // ---- one-time per-block: swizzle w2 (f32 -> f16) into A-fragment order ----
    // A element (lane,j) of frag(mtile,kb):
    //   m = mtile*16 + (lane&15),  k = kb*32 + 16*(lane>>4) + j
    {
        const int base = tid * 64;                 // 128 threads * 64 = 8192 halves
        #pragma unroll 4
        for (int i = 0; i < 64; ++i) {
            int idx   = base + i;
            int frag  = idx >> 9;                  // /512
            int lane  = (idx >> 4) & 31;
            int j     = idx & 15;
            int mtile = frag >> 1;
            int kb    = frag & 1;
            int m     = (mtile << 4) + (lane & 15);
            int k     = (kb << 5) + (((lane >> 4)) << 4) + j;
            lds_a[idx] = (_Float16)w2[k * D_MODEL + m];
        }
        if (tid < HID) { lds_w1[tid] = w1[tid]; lds_b1[tid] = b1[tid]; }
        lds_b2[tid] = b2[tid];                     // BLOCK_THREADS == D_MODEL
    }
    __syncthreads();

    const int lane   = tid & 31;
    const int waveId = tid >> 5;
    const int laneHi = lane >> 4;
    const int laneLo = lane & 15;
    const int gw     = blockIdx.x * WAVES_PER_BLOCK + waveId;

    // B layout: lane l, element j -> k = kb*32 + 16*laneHi + j. Hoist the two
    // contiguous 16-float slices of w1/b1 this lane ever needs.
    float w1lo[16], w1hi[16], b1lo[16], b1hi[16];
    #pragma unroll
    for (int j = 0; j < 16; ++j) {
        const int k0 = (laneHi << 4) + j;          // kb = 0
        w1lo[j] = lds_w1[k0];      b1lo[j] = lds_b1[k0];
        w1hi[j] = lds_w1[k0 + 32]; b1hi[j] = lds_b1[k0 + 32];
    }

    for (int it = 0; it < tiles_per_wave; ++it) {
        const int tile = gw * tiles_per_wave + it;   // wave-uniform
        if (tile >= num_tiles) break;                // wave-uniform exit
        const int tbase = tile << 4;

        // ---- per-token front-end (lane owns token laneLo; dup across halves) ----
        const int   tok  = tbase + laneLo;
        const int   vid  = variate_ids[tok];
        const int   vtyp = variate_type[vid];
        const float mu   = numeric_means[vid];
        const float sg   = numeric_stds[vid];
        const float v    = (value_num[tok] - mu) / sg;
        const int   cid  = cat_ids[tok];
        const bool  isNum = (vtyp == 0);
        const bool  isCat = (vtyp == 1) && (cid >= 0);
        const int   cidc  = cid < 0 ? 0 : cid;       // clamped, like the reference
        const float* catrow = cat_embed + (long long)cidc * D_MODEL;
        float*       outrow = out + (long long)tok * D_MODEL;

        // ---- B fragments: h = relu(v*w1 + b1) in WMMA B layout (f16) ----
        v16h bf0, bf1;
        #pragma unroll
        for (int j = 0; j < 16; ++j) {
            bf0[j] = (_Float16)fmaxf(fmaf(v, w1lo[j], b1lo[j]), 0.0f);
            bf1[j] = (_Float16)fmaxf(fmaf(v, w1hi[j], b1hi[j]), 0.0f);
        }

        // ---- 8 mtiles x 2 K-steps; acc seeded with b2; fused branchless epilogue
        #pragma unroll
        for (int mt = 0; mt < MTILES; ++mt) {
            const v16h a0 = *(const v16h*)&lds_a[(((mt << 1) + 0) * 32 + lane) << 4];
            const v16h a1 = *(const v16h*)&lds_a[(((mt << 1) + 1) * 32 + lane) << 4];
            const int  mb = (mt << 4) + (laneHi << 3);   // 8 consecutive m values

            const v4f bb0 = *(const v4f*)&lds_b2[mb];
            const v4f bb1 = *(const v4f*)&lds_b2[mb + 4];
            v8f acc;
            acc[0] = bb0.x; acc[1] = bb0.y; acc[2] = bb0.z; acc[3] = bb0.w;
            acc[4] = bb1.x; acc[5] = bb1.y; acc[6] = bb1.z; acc[7] = bb1.w;

            acc = __builtin_amdgcn_wmma_f32_16x16x32_f16(
                false, a0, false, bf0, (short)0, acc, false, false);
            acc = __builtin_amdgcn_wmma_f32_16x16x32_f16(
                false, a1, false, bf1, (short)0, acc, false, false);

            // unconditional L2-resident gather (reference computes densely too)
            const v4f c0 = *(const v4f*)(catrow + mb);
            const v4f c1 = *(const v4f*)(catrow + mb + 4);

            v4f o0, o1;
            o0.x = isNum ? acc[0] : (isCat ? c0.x : 0.0f);
            o0.y = isNum ? acc[1] : (isCat ? c0.y : 0.0f);
            o0.z = isNum ? acc[2] : (isCat ? c0.z : 0.0f);
            o0.w = isNum ? acc[3] : (isCat ? c0.w : 0.0f);
            o1.x = isNum ? acc[4] : (isCat ? c1.x : 0.0f);
            o1.y = isNum ? acc[5] : (isCat ? c1.y : 0.0f);
            o1.z = isNum ? acc[6] : (isCat ? c1.z : 0.0f);
            o1.w = isNum ? acc[7] : (isCat ? c1.w : 0.0f);

            *(v4f*)(outrow + mb)     = o0;
            *(v4f*)(outrow + mb + 4) = o1;
        }
    }
}

extern "C" void kernel_launch(void* const* d_in, const int* in_sizes, int n_in,
                              void* d_out, int out_size, void* d_ws, size_t ws_size,
                              hipStream_t stream) {
    const int*   variate_ids   = (const int*)  d_in[0];
    const float* value_num     = (const float*)d_in[1];
    const int*   cat_ids       = (const int*)  d_in[2];
    const int*   variate_type  = (const int*)  d_in[3];
    const float* numeric_means = (const float*)d_in[4];
    const float* numeric_stds  = (const float*)d_in[5];
    const float* w1            = (const float*)d_in[6];
    const float* b1            = (const float*)d_in[7];
    const float* w2            = (const float*)d_in[8];
    const float* b2            = (const float*)d_in[9];
    const float* cat_embed     = (const float*)d_in[10];
    float*       out           = (float*)d_out;

    const int total_tokens   = in_sizes[0];          // B*T
    const int num_tiles      = total_tokens / 16;    // 16 tokens per wave tile
    const int tiles_per_wave = 8;                    // amortize per-block w2 swizzle
    const int waves_needed   = (num_tiles + tiles_per_wave - 1) / tiles_per_wave;
    const int blocks         = (waves_needed + WAVES_PER_BLOCK - 1) / WAVES_PER_BLOCK;

    EventValueEmbedding_34102040330711_kernel<<<blocks, BLOCK_THREADS, 0, stream>>>(
        variate_ids, value_num, cat_ids, variate_type, numeric_means, numeric_stds,
        w1, b1, w2, b2, cat_embed, out, num_tiles, tiles_per_wave);
}
